// LSTMStream_28200755265721
// MI455X (gfx1250) — matrix operand order repaired and checked by reading, not compiled, not run on hardware
//
#include <hip/hip_runtime.h>

// LSTM: B=4096, T=512, I=10, H=20, gates=80 (i,f,g,o)
// One wave32 owns a 16-row batch tile for the full T=512 recurrence.
// gates = [x_t | h_{t-1}] (16x30, K padded to 32, f16) @ Wcat^T + bias  via
// 5x v_wmma_f32_16x16x32_f16 per timestep (bias folded into C operand).
// No cross-wave sync; x loads software-pipelined one step ahead.
// Global accesses use scalar-base + constant per-lane offset (GVS mode).

#define T_LEN 512
#define IN_SZ 10
#define HID   20
#define GATES 80
#define WAVES 4
#define RS_A  40   // A-staging row stride in halves (80 B, bank-spread, 16B aligned)
#define RS_G  84   // gate-dump row stride in floats

typedef __attribute__((ext_vector_type(16))) _Float16 v16h;
typedef __attribute__((ext_vector_type(8)))  _Float16 v8h;
typedef __attribute__((ext_vector_type(8)))  float    v8f;

// Exactly: v_mul + v_exp_f32 + v_add + v_rcp_f32 (+ v_fma for tanh)
__device__ __forceinline__ float sigm_fast(float x) {
    return __builtin_amdgcn_rcpf(1.0f + __builtin_amdgcn_exp2f(x * -1.442695041f));
}
__device__ __forceinline__ float tanh_fast(float x) {
    // tanh(x) = 2*sigmoid(2x) - 1
    return fmaf(2.0f,
        __builtin_amdgcn_rcpf(1.0f + __builtin_amdgcn_exp2f(x * -2.885390082f)),
        -1.0f);
}

__global__ __launch_bounds__(WAVES * 32)
void lstm_wmma_kernel(const float* __restrict__ x,
                      const float* __restrict__ w_ih,
                      const float* __restrict__ w_hh,
                      const float* __restrict__ b_ih,
                      const float* __restrict__ b_hh,
                      float* __restrict__ out)
{
    // Per-wave private LDS regions (no inter-wave sharing -> no barriers).
    __shared__ __align__(16) _Float16 Ast[WAVES][16][RS_A]; // [x_t|h] f16, A layout staging
    __shared__ float                  Gt [WAVES][16][RS_G]; // gate dump (f32, bias included)

    const int lane = threadIdx.x & 31;
    const int w    = threadIdx.x >> 5;
    const int tile = blockIdx.x * WAVES + w;
    const int b0   = tile * 16;

    // ---- one-time setup -------------------------------------------------
    { // zero A staging: h part must be 0 for t=0; K=30..31 stay 0 forever
        _Float16* p = &Ast[w][0][0];
        for (int i = lane; i < 16 * RS_A; i += 32) p[i] = (_Float16)0.0f;
    }

    const int nn    = lane & 15;
    const int kbase = (lane < 16) ? 0 : 16;

    // Loop-invariant weight fragments, B fragments, contiguous k order (same as A):
    // lanes 0-15: N=lane,   K=0..15 ; lanes 16-31: N=lane-16, K=16..31
    v16h bfrag[5];
    // Bias folded into WMMA C operand: C/D layout has N=lane (same col for all
    // 8 M-rows), so each lane's C vector is its gate column's bias, replicated.
    v8f cbias[5];
#pragma unroll
    for (int n = 0; n < 5; ++n) {
        v16h bf;
        const int g = n * 16 + nn;           // gate column 0..79
#pragma unroll
        for (int e = 0; e < 16; ++e) {
            const int K = kbase + e;         // concat-K: [0,10)=x, [10,30)=h
            float wv = 0.0f;
            if (K < IN_SZ)               wv = w_ih[g * IN_SZ + K];
            else if (K < IN_SZ + HID)    wv = w_hh[g * HID + (K - IN_SZ)];
            bf[e] = (_Float16)wv;
        }
        bfrag[n] = bf;
        const float bv = b_ih[g] + b_hh[g];
        v8f cb;
#pragma unroll
        for (int r = 0; r < 8; ++r) cb[r] = bv;
        cbias[n] = cb;
    }

    // Per-lane fixed assignments as CONSTANT byte offsets from a uniform base
    // (lets the compiler use saddr + voffset global addressing, no per-step
    //  64-bit vector address math).
    const char* xbase = (const char*)(x + (size_t)b0 * T_LEN * IN_SZ);
    int xoffB[5]; int xr[5], xc[5];
#pragma unroll
    for (int q = 0; q < 5; ++q) {
        const int idx = q * 32 + lane;       // 160 x values/step, 5 per lane
        xr[q] = idx / IN_SZ; xc[q] = idx % IN_SZ;
        xoffB[q] = (xr[q] * T_LEN * IN_SZ + xc[q]) * 4;
    }
    char* obase = (char*)(out + (size_t)b0 * T_LEN * HID);
    float creg[10]; int ooffB[10], em[10], ej[10];
#pragma unroll
    for (int q = 0; q < 10; ++q) {
        const int idx = q * 32 + lane;       // 320 (m,j) pairs/step, 10 per lane
        em[q] = idx / HID; ej[q] = idx % HID;
        ooffB[q] = (em[q] * T_LEN * HID + ej[q]) * 4;
        creg[q] = 0.0f;
    }

    // A-fragment read addressing (contiguous k order, same as B):
    // lanes 0-15: M=lane, K=0..15; lanes 16-31: M=lane-16, K=16..31
    const int m    = lane & 15;
    const int koff = (lane < 16) ? 0 : 16;
    const _Float16* arow = &Ast[w][m][0];
    const int mb   = (lane < 16) ? 0 : 8;    // C/D row base for this lane half

    // preload x_0
    float xv[5];
#pragma unroll
    for (int q = 0; q < 5; ++q)
        xv[q] = *(const float*)(xbase + xoffB[q]);

    // ---- recurrence -----------------------------------------------------
    for (int t = 0; t < T_LEN; ++t) {
        // stage x_t (f32 -> f16) into A staging, K = 0..9
#pragma unroll
        for (int q = 0; q < 5; ++q)
            Ast[w][xr[q]][xc[q]] = (_Float16)xv[q];
        __builtin_amdgcn_sched_barrier(0);
        __builtin_amdgcn_wave_barrier();   // same-wave LDS is in-order

        // gather A fragment: two 16-byte ds loads per lane
        v8h lo = *(const v8h*)(arow + koff);
        v8h hi = *(const v8h*)(arow + koff + 8);
        v16h a;
#pragma unroll
        for (int e = 0; e < 8; ++e) { a[e] = lo[e]; a[e + 8] = hi[e]; }

        // gates(16x80) = A(16x32) x Wcat^T + bias : 5 WMMA, f32 accum
#pragma unroll
        for (int n = 0; n < 5; ++n) {
            v8f acc = __builtin_amdgcn_wmma_f32_16x16x32_f16(
                false, a, false, bfrag[n], (short)0, cbias[n], false, false);
#pragma unroll
            for (int r = 0; r < 8; ++r)
                Gt[w][mb + r][n * 16 + nn] = acc[r];
        }

        // prefetch x_{t+1}: uniform base advance + constant lane offsets;
        // global latency hides behind the elementwise phase
        const int tn = (t + 1 < T_LEN) ? (t + 1) : t;
        const char* xbt = xbase + (size_t)tn * (IN_SZ * 4);
#pragma unroll
        for (int q = 0; q < 5; ++q)
            xv[q] = *(const float*)(xbt + xoffB[q]);

        __builtin_amdgcn_sched_barrier(0);
        __builtin_amdgcn_wave_barrier();

        // elementwise LSTM cell on 10 (m,j) pairs per lane
        char* obt = obase + (size_t)t * (HID * 4);
#pragma unroll
        for (int q = 0; q < 10; ++q) {
            const int mm = em[q], j = ej[q];
            const float ig = sigm_fast(Gt[w][mm][j]);
            const float fg = sigm_fast(Gt[w][mm][20 + j]);
            const float gg = tanh_fast(Gt[w][mm][40 + j]);
            const float og = sigm_fast(Gt[w][mm][60 + j]);
            const float c  = fg * creg[q] + ig * gg;
            creg[q] = c;
            const float h  = og * tanh_fast(c);
            *(float*)(obt + ooffB[q]) = h;              // out[b, t, j] (f32)
            Ast[w][mm][IN_SZ + j] = (_Float16)h;        // h for next step's A
        }
        __builtin_amdgcn_sched_barrier(0);
        __builtin_amdgcn_wave_barrier();
    }
}

extern "C" void kernel_launch(void* const* d_in, const int* in_sizes, int n_in,
                              void* d_out, int out_size, void* d_ws, size_t ws_size,
                              hipStream_t stream) {
    const float* x    = (const float*)d_in[0];  // [4096, 512, 10]
    const float* w_ih = (const float*)d_in[1];  // [80, 10]
    const float* w_hh = (const float*)d_in[2];  // [80, 20]
    const float* b_ih = (const float*)d_in[3];  // [80]
    const float* b_hh = (const float*)d_in[4];  // [80]
    float* out = (float*)d_out;                 // [4096, 512, 20]

    // 256 batch tiles of 16 rows; 4 waves per block -> 64 blocks
    lstm_wmma_kernel<<<dim3(64), dim3(WAVES * 32), 0, stream>>>(
        x, w_ih, w_hh, b_ih, b_hh, out);
}
